// FSAS_24953759990225
// MI455X (gfx1250) — matrix operand order repaired and checked by reading, not compiled, not run on hardware
//
#include <hip/hip_runtime.h>

// MI455X (gfx1250) fully-fused kernel for the frequency-domain attention block.
// One workgroup (256 threads = 8 wave32) per 8x8 output patch per batch image.
// All intermediates live in LDS (~232 KB of the WGP's 320 KB); HBM traffic is
// just x in (64 MB) + out (64 MB). GEMMs run on v_wmma_f32_16x16x32_bf16 with
// fragment-native LDS layouts (v16bf loads -> ds_load_b128), and the depthwise
// weights arrive via the Tensor Data Mover (tensor_load_to_lds + tensorcnt).

typedef __bf16 bf16;
typedef __attribute__((ext_vector_type(16))) __bf16 v16bf;
typedef __attribute__((ext_vector_type(8)))  float  v8f;
typedef __attribute__((ext_vector_type(4))) unsigned int u32x4;
typedef __attribute__((ext_vector_type(8))) int i32x8;
typedef __attribute__((ext_vector_type(4))) int i32x4;

#define CIN   64     // input channels (d)
#define D2    128    // 2*d  (q/k/v channel count)
#define HW    256    // H == W
#define NPOS  100    // 10x10 halo positions
#define NPAD  112    // padded to 7 WMMA N-tiles of 16

// Row strides (elements). All chosen as odd multiples of 8 DWORDs for LDS bank
// spread while keeping v16bf (32B) fragment loads aligned.
#define XS_STR  80   // x_s   [112][80]  bf16, [pos][cin]
#define WS_STR  80   // w_s   [128][80]  bf16, [oc][cin]
#define WOS_STR 144  // wo_s  [64][144]  bf16, [oc][ch]
#define GS_STR  144  // g_s   [64][144]  bf16, [pos][ch]

// bf16: 112*80 + 64*144 + 128*80 + 64*144 = 37632 elems = 75264 B
// f32 : 128*9 + 128*112 + 3*128*64 + 384  = 40448 elems = 161792 B
#define SMEM_BYTES (75264 + 161792)   // 237056 B

__device__ __forceinline__ bf16 tobf(float f) { return (bf16)f; }

// K-block permutation for A fragments: within each 32-K group store 8-element
// blocks in order [0,2,1,3] so that fragment (k0,lhi) is 16 contiguous bf16 at
// offset k0 + lhi*16 (matches ISA 7.12.2: elems 0..7 = K base..+7, 8..15 = +16..+23).
__device__ __forceinline__ int kperm(int c) {
  const int bl = c >> 3;
  const int sb = (bl & ~3) | ((bl & 1) << 1) | ((bl >> 1) & 1);
  return (c & 7) | (sb << 3);
}

// 32 outputs of an 8x8 circular 2D convolution; BASE is compile-time so every
// q/k index folds and the tiles stay in VGPRs (no scratch).
template <int BASE>
__device__ __forceinline__ void circ32(const float* qv, const float* kv, float* orow) {
#pragma unroll
  for (int pp = 0; pp < 32; ++pp) {
    const int p = BASE + pp, r = p >> 3, c = p & 7;
    float s = 0.f;
#pragma unroll
    for (int i = 0; i < 8; ++i)
#pragma unroll
      for (int j = 0; j < 8; ++j)
        s = fmaf(qv[i * 8 + j], kv[((r - i) & 7) * 8 + ((c - j) & 7)], s);
    orow[p] = s;
  }
}

// 1-D contiguous fp32 tile via the Tensor Data Mover (D# built per ISA 8.3/8.4).
__device__ __forceinline__ void tdm_load_1d(unsigned lds_off, const void* gptr, unsigned n) {
  const unsigned long long ga = (unsigned long long)(__UINTPTR_TYPE__)gptr;
  u32x4 g0 = {};
  g0[0] = 1u;                                             // count=1 (valid), user mode
  g0[1] = lds_off;                                        // lds_addr (bytes)
  g0[2] = (unsigned)(ga & 0xffffffffu);                   // global_addr[31:0]
  g0[3] = (unsigned)((ga >> 32) & 0x01ffffffu)            // global_addr[56:32]
        | (2u << 30);                                     // type = 2 ("image")
  i32x8 g1 = {};
  g1[0] = (int)(2u << 16);                                // data_size = 4B, no multicast
  g1[1] = (int)((n & 0xffffu) << 16);                     // tensor_dim0[15:0]
  g1[2] = (int)(n >> 16);                                 // tensor_dim0[31:16]
  g1[3] = (int)((n & 0xffffu) << 16);                     // tile_dim0
  g1[4] = 1;                                              // tile_dim1 = 1
  g1[5] = (int)n;                                         // tensor_dim0_stride
  const i32x4 z = {};
#if defined(__clang_major__) && __clang_major__ >= 23
  const i32x8 z8 = {};
  __builtin_amdgcn_tensor_load_to_lds(g0, g1, z, z, z8, 0);
#else
  __builtin_amdgcn_tensor_load_to_lds(g0, g1, z, z, 0);
#endif
}

__global__ __launch_bounds__(256)
void fsas_fused(const float* __restrict__ x,   const float* __restrict__ w1,
                const float* __restrict__ wdw, const float* __restrict__ wout,
                const float* __restrict__ lnw, const float* __restrict__ lnb,
                float* __restrict__ out) {
  extern __shared__ char smem[];
  bf16*  x_s   = (bf16*)smem;              // [112][80]  input halo, [pos][cin]
  bf16*  wo_s  = x_s  + NPAD * XS_STR;     // [64][144]  project_out weights (A, [oc][ch])
  bf16*  w_s   = wo_s + 64 * WOS_STR;      // [128][80]  per-group 1x1 weights (A, [oc][cin])
  bf16*  g_s   = w_s  + D2 * WS_STR;       // [64][144]  gated activations (B, [pos][ch])
  float* wdw_s = (float*)(g_s + 64 * GS_STR); // [128][9] depthwise weights (TDM target)
  float* h_s   = wdw_s + D2 * 9;           // [128][112] hidden (reused as o_s[128][64])
  float* qd    = h_s + D2 * NPAD;          // [128][64]
  float* kd    = qd  + D2 * 64;            // [128][64]
  float* vd    = kd  + D2 * 64;            // [128][64]
  float* lnw_s = vd  + D2 * 64;            // [128]
  float* lnb_s = lnw_s + D2;               // [128]
  float* mu_s  = lnb_s + D2;               // [64]
  float* rs_s  = mu_s  + 64;               // [64]

  const int tid  = threadIdx.x;
  const int lane = tid & 31, wave = tid >> 5;
  const int lrow = lane & 15, lhi = lane >> 4;     // WMMA lane decomposition
  const int px = blockIdx.x, py = blockIdx.y, b = blockIdx.z;

  // ---- Phase 0: stage input halo (10x10, zero-padded) + static weights ----
  for (int idx = tid; idx < CIN * NPAD; idx += 256) {
    int c = idx / NPAD, pos = idx % NPAD;
    float v = 0.f;
    if (pos < NPOS) {
      int rr = pos / 10, cc = pos % 10;
      int gr = py * 8 - 1 + rr, gc = px * 8 - 1 + cc;
      if ((unsigned)gr < HW && (unsigned)gc < HW)
        v = x[(((size_t)b * CIN + c) * HW + gr) * HW + gc];
    }
    x_s[pos * XS_STR + c] = tobf(v);       // B layout: [pos][cin], K contiguous
  }
  for (int idx = tid; idx < 64 * D2; idx += 256) {
    int o = idx >> 7, c = idx & 127;
    wo_s[o * WOS_STR + c] = tobf(wout[idx]);
  }
  for (int idx = tid; idx < D2; idx += 256) { lnw_s[idx] = lnw[idx]; lnb_s[idx] = lnb[idx]; }

  // ---- Phase 1: per group (q,k,v): 1x1-conv GEMM (WMMA) + depthwise 3x3 ----
  for (int g = 0; g < 3; ++g) {
    for (int idx = tid; idx < D2 * CIN; idx += 256) {
      int o = idx >> 6, c = idx & 63;
      w_s[o * WS_STR + c] = tobf(w1[g * D2 * CIN + idx]);
    }
    if (wave == 0) {                        // TDM: depthwise weights, 1152 fp32 tile
      tdm_load_1d((unsigned)((char*)wdw_s - smem), wdw + g * D2 * 9, D2 * 9);
      __builtin_amdgcn_s_wait_tensorcnt(0);
    }
    if (g < 2) __builtin_prefetch(&w1[(g + 1) * D2 * CIN + tid * 32], 0, 1); // global_prefetch_b8
    __syncthreads();

    // hidden[128][112] = w1_g[128][64] x x[64][112]; 8x7 = 56 tiles, 7 per wave.
    for (int i = 0; i < 7; ++i) {
      const int t = wave * 7 + i, mt = t / 7, nt = t % 7;
      v8f acc = {};
#pragma unroll
      for (int k0 = 0; k0 < CIN; k0 += 32) {
        const v16bf A = *(const v16bf*)(w_s + (mt * 16 + lrow) * WS_STR + k0 + lhi * 16);
        const v16bf B = *(const v16bf*)(x_s + (nt * 16 + lrow) * XS_STR + k0 + lhi * 16);
        acc = __builtin_amdgcn_wmma_f32_16x16x32_bf16(false, A, false, B,
                                                      (short)0, acc, false, false);
      }
      // C/D: N = lane%16, M = vgpr + 8*(lane/16)
#pragma unroll
      for (int r = 0; r < 8; ++r)
        h_s[(mt * 16 + lhi * 8 + r) * NPAD + nt * 16 + lrow] = acc[r];
    }
    __syncthreads();

    // depthwise 3x3 (SAME): halo rows/cols -1..8 are h_s positions 0..9
    float* dst = (g == 0) ? qd : (g == 1) ? kd : vd;
    for (int idx = tid; idx < D2 * 64; idx += 256) {
      const int ch = idx >> 6, p = idx & 63, r = p >> 3, c = p & 7;
      const float* hrow = h_s + ch * NPAD;
      const float* wk   = wdw_s + ch * 9;
      float a = 0.f;
#pragma unroll
      for (int dr = 0; dr < 3; ++dr)
#pragma unroll
        for (int dc = 0; dc < 3; ++dc)
          a = fmaf(hrow[(r + dr) * 10 + (c + dc)], wk[dr * 3 + dc], a);
      dst[idx] = a;
    }
    __syncthreads();
  }

  // ---- Phase 2: irfft2(rfft2(q)*rfft2(k)) == 8x8 circular convolution ----
  {
    const int ch = tid & 127, half = tid >> 7;   // wave-uniform split
    float qv[64], kv[64];
    const float4* q4 = (const float4*)(qd + ch * 64);
    const float4* k4 = (const float4*)(kd + ch * 64);
#pragma unroll
    for (int i = 0; i < 16; ++i) {               // ds_load_b128 register-tile fills
      float4 tq = q4[i], tk = k4[i];
      qv[4*i+0] = tq.x; qv[4*i+1] = tq.y; qv[4*i+2] = tq.z; qv[4*i+3] = tq.w;
      kv[4*i+0] = tk.x; kv[4*i+1] = tk.y; kv[4*i+2] = tk.z; kv[4*i+3] = tk.w;
    }
    float* o_s = h_s;                            // reuse hidden buffer
    if (half == 0) circ32<0>(qv, kv, o_s + ch * 64);
    else           circ32<32>(qv, kv, o_s + ch * 64);
  }
  __syncthreads();

  // ---- Phase 3: channel LayerNorm (biased var), gate by v, pack bf16 ----
  {
    float* o_s = h_s;
    if (tid < 64) {
      float s = 0.f, s2 = 0.f;
      for (int ch = 0; ch < D2; ++ch) { float v = o_s[ch * 64 + tid]; s += v; s2 += v * v; }
      const float mu  = s * (1.f / D2);
      const float var = s2 * (1.f / D2) - mu * mu;
      mu_s[tid] = mu;
      rs_s[tid] = rsqrtf(var + 1e-5f);
    }
    __syncthreads();
    for (int idx = tid; idx < D2 * 64; idx += 256) {
      const int ch = idx >> 6, p = idx & 63;
      float v = (o_s[idx] - mu_s[p]) * rs_s[p] * lnw_s[ch] + lnb_s[ch];
      g_s[p * GS_STR + ch] = tobf(v * vd[idx]);  // B layout: [pos][ch], K contiguous
    }
  }
  __syncthreads();

  // ---- Phase 4: project_out GEMM: out[64][64] = wo[64][128] x g[128][64] ----
  for (int i = 0; i < 2; ++i) {
    const int t = wave * 2 + i, mt = t >> 2, nt = t & 3;
    v8f acc = {};
#pragma unroll
    for (int k0 = 0; k0 < D2; k0 += 32) {
      const v16bf A = *(const v16bf*)(wo_s + (mt * 16 + lrow) * WOS_STR + k0 + lhi * 16);
      const v16bf B = *(const v16bf*)(g_s  + (nt * 16 + lrow) * GS_STR  + k0 + lhi * 16);
      acc = __builtin_amdgcn_wmma_f32_16x16x32_bf16(false, A, false, B,
                                                    (short)0, acc, false, false);
    }
    const int pos = nt * 16 + lrow, pr = pos >> 3, pc = pos & 7;
#pragma unroll
    for (int r = 0; r < 8; ++r) {
      const int oc = mt * 16 + lhi * 8 + r;
      out[(((size_t)b * 64 + oc) * HW + py * 8 + pr) * HW + px * 8 + pc] = acc[r];
    }
  }
}

extern "C" void kernel_launch(void* const* d_in, const int* in_sizes, int n_in,
                              void* d_out, int out_size, void* d_ws, size_t ws_size,
                              hipStream_t stream) {
  (void)in_sizes; (void)n_in; (void)out_size; (void)d_ws; (void)ws_size;
  const float* x    = (const float*)d_in[0];
  const float* w1   = (const float*)d_in[1];
  const float* wdw  = (const float*)d_in[2];
  const float* wout = (const float*)d_in[3];
  const float* lnw  = (const float*)d_in[4];
  const float* lnb  = (const float*)d_in[5];
  float* out = (float*)d_out;

  // ~232 KB dynamic LDS > default cap: raise the attribute (deterministic, no stream op).
  hipFuncSetAttribute((const void*)fsas_fused,
                      hipFuncAttributeMaxDynamicSharedMemorySize, SMEM_BYTES);

  dim3 grid(HW / 8, HW / 8, 4);   // 32 x 32 patches x B=4
  fsas_fused<<<grid, 256, SMEM_BYTES, stream>>>(x, w1, wdw, wout, lnw, lnb, out);
}
